// SelfAttention_42245298323823
// MI455X (gfx1250) — matrix operand order repaired and checked by reading, not compiled, not run on hardware
//
#include <hip/hip_runtime.h>
#include <hip/hip_bf16.h>

// ---------------------------------------------------------------------------
// Types for CDNA5 WMMA (wave32): D(16x16 f32) = A(16x32 bf16) * B(32x16 bf16) + C
// ---------------------------------------------------------------------------
typedef __attribute__((ext_vector_type(16))) __bf16 v16bf;
typedef __attribute__((ext_vector_type(8)))  float  v8f;
typedef int vint4 __attribute__((vector_size(16)));   // matches builtin param type

union BFrag {
    unsigned int u[8];
    v16bf v;
};

#define AS1 __attribute__((address_space(1)))
#define AS3 __attribute__((address_space(3)))

// ---- async global->LDS (CDNA5 GLOBAL_LOAD_ASYNC_TO_LDS_B128, ASYNCcnt) ----
#if __has_builtin(__builtin_amdgcn_global_load_async_to_lds_b128)
#define USE_ASYNC 1
__device__ __forceinline__ void async_b128(const void* gsrc, void* ldst) {
    __builtin_amdgcn_global_load_async_to_lds_b128(
        (AS1 vint4*)gsrc, (AS3 vint4*)ldst, 0, 0);
}
__device__ __forceinline__ void async_wait0() {
#if __has_builtin(__builtin_amdgcn_s_wait_asynccnt)
    __builtin_amdgcn_s_wait_asynccnt(0);
#else
    asm volatile("s_wait_asynccnt 0x0" ::: "memory");
#endif
}
#else
#define USE_ASYNC 0
#endif

__device__ __forceinline__ unsigned short f2bf(float x) {
    unsigned int u = __float_as_uint(x);
    unsigned int r = (u + 0x7FFFu + ((u >> 16) & 1u)) >> 16;   // RNE
    return (unsigned short)r;
}
__device__ __forceinline__ unsigned int pack2bf(float a, float b) {
    return (unsigned int)f2bf(a) | ((unsigned int)f2bf(b) << 16);
}

// A fragment: 16x32 bf16, rows row0..row0+15, K cols k0..k0+31 from LDS (row-major, ld elems).
// Contiguous k order used here (matches load_b): lane l<16 -> row l, K=0..15 (pairs/VGPR); lane l+16 -> K=16..31.
__device__ __forceinline__ BFrag load_a(const unsigned short* p0, int ld, int row0, int k0) {
    const int lane = threadIdx.x & 31;
    const int r  = row0 + (lane & 15);
    const int kh = (lane >> 4) * 16;
    const unsigned short* p = p0 + r * ld + k0;
    BFrag f;
#pragma unroll
    for (int i = 0; i < 8; ++i) f.u[i] = *(const unsigned int*)(p + kh + 2 * i);
    return f;
}

// B fragment: 32x16 bf16 (KxN), from LDS holding B-transposed [n][k] row-major (ld elems).
// Contiguous k order used here (matches load_a): lane n<16 -> col n, K=0..15 (pairs/VGPR); lane n+16 -> K=16..31.
__device__ __forceinline__ BFrag load_b(const unsigned short* p0, int ld, int n0, int k0) {
    const int lane = threadIdx.x & 31;
    const int n  = n0 + (lane & 15);
    const int kh = (lane >> 4) * 16;
    const unsigned short* p = p0 + n * ld + k0 + kh;
    BFrag f;
#pragma unroll
    for (int i = 0; i < 8; ++i) f.u[i] = *(const unsigned int*)(p + 2 * i);
    return f;
}

__device__ __forceinline__ v8f wmma_bf16(const BFrag& a, const BFrag& b, v8f c) {
    return __builtin_amdgcn_wmma_f32_16x16x32_bf16(false, a.v, false, b.v,
                                                   (short)0, c, false, false);
}

// Problem constants
#define BATCH  2
#define SEQ    2048
#define HID    2048
#define NH     16
#define HD     128
#define N3H    6144
#define MROWS  4096          // BATCH*SEQ
#define LOG_RB_OVER_HALF 0.14391157f   // ln(10000)/64

// ---------------------------------------------------------------------------
// Kernel 1: qkv = hs @ Wqkv + b ; RoPE(q,k) ; write Q,K bf16 [bh][s][d], V^T bf16 [bh][d][s]
// Grid: (48 n-tiles, 32 m-tiles), 256 threads (8 waves), tile 128x128, k-step 32.
// ---------------------------------------------------------------------------
__global__ __launch_bounds__(256) void k1_qkv_rope(
    const float* __restrict__ hs, const float* __restrict__ W,
    const float* __restrict__ bias,
    unsigned short* __restrict__ Qg, unsigned short* __restrict__ Kg,
    unsigned short* __restrict__ Vtg)
{
    __shared__ __align__(16) unsigned short sA[128 * 36];
    __shared__ __align__(16) unsigned short sB[128 * 36];

    const int tid = threadIdx.x, lane = tid & 31, wave = tid >> 5;
    const int half = lane >> 4, l15 = lane & 15;
    const int n0 = blockIdx.x * 128, m0 = blockIdx.y * 128;

    v8f acc[8];
#pragma unroll
    for (int i = 0; i < 8; ++i)
#pragma unroll
        for (int j = 0; j < 8; ++j) acc[i][j] = 0.f;

    for (int k0 = 0; k0 < HID; k0 += 32) {
        __syncthreads();
        // stage A tile (128 x 32 fp32 -> bf16)
#pragma unroll
        for (int it = 0; it < 4; ++it) {
            int lin = tid + it * 256;
            int r = lin >> 3, c4 = lin & 7;
            float4 f = *(const float4*)(hs + (size_t)(m0 + r) * HID + k0 + c4 * 4);
            *(unsigned int*)(sA + r * 36 + c4 * 4)     = pack2bf(f.x, f.y);
            *(unsigned int*)(sA + r * 36 + c4 * 4 + 2) = pack2bf(f.z, f.w);
        }
        // stage B tile transposed: Wqkv[k0..+31][n0..+127] -> sB[n][k] bf16
#pragma unroll
        for (int it = 0; it < 4; ++it) {
            int lin = tid + it * 256;
            int kr = lin >> 5, c4 = lin & 31;
            float4 f = *(const float4*)(W + (size_t)(k0 + kr) * N3H + n0 + c4 * 4);
            sB[(c4 * 4 + 0) * 36 + kr] = f2bf(f.x);
            sB[(c4 * 4 + 1) * 36 + kr] = f2bf(f.y);
            sB[(c4 * 4 + 2) * 36 + kr] = f2bf(f.z);
            sB[(c4 * 4 + 3) * 36 + kr] = f2bf(f.w);
        }
        if (k0 + 32 < HID)
            __builtin_prefetch(hs + (size_t)(m0 + (tid >> 1)) * HID + k0 + 32, 0, 1);
        __syncthreads();

        BFrag af = load_a(sA, 36, wave * 16, 0);
#pragma unroll
        for (int di = 0; di < 8; ++di) {
            BFrag bf = load_b(sB, 36, di * 16, 0);
            acc[di] = wmma_bf16(af, bf, acc[di]);
        }
    }

    // bias
#pragma unroll
    for (int di = 0; di < 8; ++di) {
        float bv = bias[n0 + di * 16 + l15];
#pragma unroll
        for (int j = 0; j < 8; ++j) acc[di][j] += bv;
    }

    const int rowbase = m0 + wave * 16;
    const int sec = n0 >> 11;                // 0=q, 1=k, 2=v (2048-wide sections)
    const int h = (n0 & 2047) >> 7;          // head index (tile spans exactly one head)

    if (sec == 2) {
        // V: store transposed bf16  Vt[bh][d][s]
#pragma unroll
        for (int di = 0; di < 8; ++di) {
            int d = di * 16 + l15;
#pragma unroll
            for (int j = 0; j < 8; ++j) {
                int gr = rowbase + j + 8 * half;
                int s = gr & (SEQ - 1), b = gr >> 11, bh = b * NH + h;
                Vtg[((size_t)bh * HD + d) * SEQ + s] = f2bf(acc[di][j]);
            }
        }
    } else {
        unsigned short* G = (sec == 0) ? Qg : Kg;
        // RoPE: pair (d, d+64) == (acc[di], acc[di+4]) at same lane, di<4
#pragma unroll
        for (int di = 0; di < 4; ++di) {
            int jf = di * 16 + l15;          // freq index == low-half d
            float invf = __expf(-(float)jf * LOG_RB_OVER_HALF);
#pragma unroll
            for (int j = 0; j < 8; ++j) {
                int gr = rowbase + j + 8 * half;
                int s = gr & (SEQ - 1), b = gr >> 11, bh = b * NH + h;
                float x = acc[di][j], y = acc[di + 4][j];
                float ang = (float)s * invf;
                float sn, cs;
                __sincosf(ang, &sn, &cs);
                unsigned short* base = G + ((size_t)bh * SEQ + s) * HD + jf;
                base[0]  = f2bf(x * cs - y * sn);
                base[64] = f2bf(y * cs + x * sn);
            }
        }
    }
}

// ---------------------------------------------------------------------------
// Kernel 2: causal flash attention. Grid (32 q-tiles, 32 bh), 128 threads (4 waves).
// Q tile 64x128, K tile 64x128, V^T tile 128x64, per-wave 16-row strip.
// K/V/Q tiles staged via GLOBAL_LOAD_ASYNC_TO_LDS_B128 when available.
// ---------------------------------------------------------------------------
__global__ __launch_bounds__(128) void k2_attn(
    const unsigned short* __restrict__ Qg, const unsigned short* __restrict__ Kg,
    const unsigned short* __restrict__ Vtg, unsigned short* __restrict__ AOg)
{
    __shared__ __align__(16) unsigned short Qs[64 * 128];
    __shared__ __align__(16) unsigned short Ks[64 * 128];
    __shared__ __align__(16) unsigned short Vts[128 * 64];
    __shared__ __align__(16) unsigned short Ps[4 * 16 * 64];

    const int tid = threadIdx.x, lane = tid & 31, wave = tid >> 5;
    const int half = lane >> 4, l15 = lane & 15;
    const int qt = blockIdx.x, bh = blockIdx.y;
    const int b = bh >> 4, h = bh & 15;
    const float scale = 0.08838834764831845f;   // 1/sqrt(128)

    // load Q tile (64 rows x 128 bf16)
#pragma unroll
    for (int it = 0; it < 8; ++it) {
        int lin = tid + it * 128;
        int r = lin >> 4, c = lin & 15;
#if USE_ASYNC
        async_b128(Qg + ((size_t)bh * SEQ + qt * 64 + r) * HD + c * 8,
                   Qs + r * 128 + c * 8);
#else
        ((uint4*)(Qs + r * 128))[c] =
            ((const uint4*)(Qg + ((size_t)bh * SEQ + qt * 64 + r) * HD))[c];
#endif
    }

    v8f o[8];
    float m[8], l[8];
#pragma unroll
    for (int i = 0; i < 8; ++i) {
        m[i] = -1e30f; l[i] = 0.f;
#pragma unroll
        for (int j = 0; j < 8; ++j) o[i][j] = 0.f;
    }

    for (int kt = 0; kt <= qt; ++kt) {
        __syncthreads();
        // K tile (64 x 128) and V^T tile (128 x 64)
#pragma unroll
        for (int it = 0; it < 8; ++it) {
            int lin = tid + it * 128;
            int r = lin >> 4, c = lin & 15;
            int d = lin >> 3, cv = lin & 7;
#if USE_ASYNC
            async_b128(Kg + ((size_t)bh * SEQ + kt * 64 + r) * HD + c * 8,
                       Ks + r * 128 + c * 8);
            async_b128(Vtg + ((size_t)bh * HD + d) * SEQ + kt * 64 + cv * 8,
                       Vts + d * 64 + cv * 8);
#else
            ((uint4*)(Ks + r * 128))[c] =
                ((const uint4*)(Kg + ((size_t)bh * SEQ + kt * 64 + r) * HD))[c];
            ((uint4*)(Vts + d * 64))[cv] =
                ((const uint4*)(Vtg + ((size_t)bh * HD + d) * SEQ + kt * 64))[cv];
#endif
        }
#if USE_ASYNC
        async_wait0();
#endif
        __syncthreads();

        // S = Q K^T   (16x64 strip per wave)
        v8f sc[4];
#pragma unroll
        for (int c = 0; c < 4; ++c)
#pragma unroll
            for (int j = 0; j < 8; ++j) sc[c][j] = 0.f;
#pragma unroll
        for (int kk = 0; kk < 4; ++kk) {
            BFrag af = load_a(Qs, 128, wave * 16, kk * 32);
#pragma unroll
            for (int c = 0; c < 4; ++c) {
                BFrag bf = load_b(Ks, 128, c * 16, kk * 32);
                sc[c] = wmma_bf16(af, bf, sc[c]);
            }
        }

        // scale + causal mask (only diagonal tile needs it)
#pragma unroll
        for (int c = 0; c < 4; ++c)
#pragma unroll
            for (int j = 0; j < 8; ++j) {
                float v = sc[c][j] * scale;
                if (kt == qt) {
                    int qr = qt * 64 + wave * 16 + j + 8 * half;
                    int ky = kt * 64 + c * 16 + l15;
                    if (ky > qr) v = -1e30f;
                }
                sc[c][j] = v;
            }

        // row max across 64 cols: 4 local + 16-lane half-wave reduce
        float mn[8];
#pragma unroll
        for (int j = 0; j < 8; ++j)
            mn[j] = fmaxf(fmaxf(sc[0][j], sc[1][j]), fmaxf(sc[2][j], sc[3][j]));
#pragma unroll
        for (int msk = 1; msk < 16; msk <<= 1)
#pragma unroll
            for (int j = 0; j < 8; ++j) mn[j] = fmaxf(mn[j], __shfl_xor(mn[j], msk, 32));

        float sf[8];
#pragma unroll
        for (int j = 0; j < 8; ++j) {
            float mnew = fmaxf(m[j], mn[j]);
            sf[j] = __expf(m[j] - mnew);
            m[j] = mnew;
        }

        // P = exp(S - m), row sums, stash P (bf16) for PV wmma
        float rs[8];
#pragma unroll
        for (int j = 0; j < 8; ++j) rs[j] = 0.f;
#pragma unroll
        for (int c = 0; c < 4; ++c)
#pragma unroll
            for (int j = 0; j < 8; ++j) {
                float p = __expf(sc[c][j] - m[j]);
                rs[j] += p;
                Ps[(wave * 16 + j + 8 * half) * 64 + c * 16 + l15] = f2bf(p);
            }
#pragma unroll
        for (int msk = 1; msk < 16; msk <<= 1)
#pragma unroll
            for (int j = 0; j < 8; ++j) rs[j] += __shfl_xor(rs[j], msk, 32);
#pragma unroll
        for (int j = 0; j < 8; ++j) l[j] = l[j] * sf[j] + rs[j];

        // rescale O, then O += P V
#pragma unroll
        for (int di = 0; di < 8; ++di)
#pragma unroll
            for (int j = 0; j < 8; ++j) o[di][j] *= sf[j];

        __syncthreads();
#pragma unroll
        for (int kk = 0; kk < 2; ++kk) {
            BFrag pf = load_a(Ps + wave * 16 * 64, 64, 0, kk * 32);
#pragma unroll
            for (int di = 0; di < 8; ++di) {
                BFrag vf = load_b(Vts, 64, di * 16, kk * 32);
                o[di] = wmma_bf16(pf, vf, o[di]);
            }
        }
    }

    // normalize + store attention output bf16, layout (B*S, H) row-major
#pragma unroll
    for (int di = 0; di < 8; ++di)
#pragma unroll
        for (int j = 0; j < 8; ++j) {
            int s = qt * 64 + wave * 16 + j + 8 * half;
            float val = o[di][j] / l[j];
            AOg[((size_t)b * SEQ + s) * HID + h * HD + di * 16 + l15] = f2bf(val);
        }
}

// ---------------------------------------------------------------------------
// Kernel 3: out = AO(bf16) @ Wproj(fp32->bf16), fp32 store.
// Grid (16 n-tiles, 32 m-tiles), 256 threads, tile 128x128, k-step 32.
// A tile staged via async LDS DMA when available (already bf16).
// ---------------------------------------------------------------------------
__global__ __launch_bounds__(256) void k3_proj(
    const unsigned short* __restrict__ A, const float* __restrict__ W,
    float* __restrict__ out)
{
    __shared__ __align__(16) unsigned short sA[128 * 40];   // ld=40: 16B-aligned rows
    __shared__ __align__(16) unsigned short sB[128 * 36];

    const int tid = threadIdx.x, lane = tid & 31, wave = tid >> 5;
    const int half = lane >> 4, l15 = lane & 15;
    const int n0 = blockIdx.x * 128, m0 = blockIdx.y * 128;

    v8f acc[8];
#pragma unroll
    for (int i = 0; i < 8; ++i)
#pragma unroll
        for (int j = 0; j < 8; ++j) acc[i][j] = 0.f;

    for (int k0 = 0; k0 < HID; k0 += 32) {
        __syncthreads();
        // A tile: 128 rows x 32 bf16 (4 x B128 per row)
#pragma unroll
        for (int it = 0; it < 2; ++it) {
            int lin = tid + it * 256;
            int r = lin >> 2, c = lin & 3;
#if USE_ASYNC
            async_b128(A + (size_t)(m0 + r) * HID + k0 + c * 8,
                       sA + r * 40 + c * 8);
#else
            *(uint4*)(sA + r * 40 + c * 8) =
                *(const uint4*)(A + (size_t)(m0 + r) * HID + k0 + c * 8);
#endif
        }
        // B tile transposed: Wproj[k0..+31][n0..+127] fp32 -> sB[n][k] bf16
#pragma unroll
        for (int it = 0; it < 4; ++it) {
            int lin = tid + it * 256;
            int kr = lin >> 5, c4 = lin & 31;
            float4 f = *(const float4*)(W + (size_t)(k0 + kr) * HID + n0 + c4 * 4);
            sB[(c4 * 4 + 0) * 36 + kr] = f2bf(f.x);
            sB[(c4 * 4 + 1) * 36 + kr] = f2bf(f.y);
            sB[(c4 * 4 + 2) * 36 + kr] = f2bf(f.z);
            sB[(c4 * 4 + 3) * 36 + kr] = f2bf(f.w);
        }
        if (k0 + 32 < HID)
            __builtin_prefetch(W + (size_t)(k0 + 32 + (tid >> 3)) * HID + n0, 0, 1);
#if USE_ASYNC
        async_wait0();
#endif
        __syncthreads();

        BFrag af = load_a(sA, 40, wave * 16, 0);
#pragma unroll
        for (int di = 0; di < 8; ++di) {
            BFrag bf = load_b(sB, 36, di * 16, 0);
            acc[di] = wmma_bf16(af, bf, acc[di]);
        }
    }

#pragma unroll
    for (int di = 0; di < 8; ++di)
#pragma unroll
        for (int j = 0; j < 8; ++j) {
            int gr = m0 + wave * 16 + j + 8 * half;
            out[(size_t)gr * HID + n0 + di * 16 + l15] = acc[di][j];
        }
}

// ---------------------------------------------------------------------------
extern "C" void kernel_launch(void* const* d_in, const int* in_sizes, int n_in,
                              void* d_out, int out_size, void* d_ws, size_t ws_size,
                              hipStream_t stream) {
    (void)in_sizes; (void)n_in; (void)out_size; (void)ws_size;
    const float* hs    = (const float*)d_in[0];   // (2,2048,2048)
    const float* Wqkv  = (const float*)d_in[1];   // (2048,6144)
    const float* bqkv  = (const float*)d_in[2];   // (6144,)
    const float* Wproj = (const float*)d_in[3];   // (2048,2048)
    float* out = (float*)d_out;                   // (2,2048,2048) fp32

    // Workspace: 4 bf16 regions of 32*2048*128 = 8Mi elements each (64 MB total)
    unsigned short* base = (unsigned short*)d_ws;
    unsigned short* Qg  = base;                            // [bh][s][d]
    unsigned short* Kg  = base + (size_t)8 * 1024 * 1024;  // [bh][s][d]
    unsigned short* Vtg = base + (size_t)16 * 1024 * 1024; // [bh][d][s]
    unsigned short* AOg = base + (size_t)24 * 1024 * 1024; // (B*S, H)

    dim3 g1(N3H / 128, MROWS / 128);   // (48, 32)
    k1_qkv_rope<<<g1, 256, 0, stream>>>(hs, Wqkv, bqkv, Qg, Kg, Vtg);

    dim3 g2(SEQ / 64, BATCH * NH);     // (32, 32)
    k2_attn<<<g2, 128, 0, stream>>>(Qg, Kg, Vtg, AOg);

    dim3 g3(HID / 128, MROWS / 128);   // (16, 32)
    k3_proj<<<g3, 256, 0, stream>>>(AOg, Wproj, out);
}
